// Plamo2MambaMixer_18837726560960
// MI455X (gfx1250) — matrix operand order repaired and checked by reading, not compiled, not run on hardware
//
#include <hip/hip_runtime.h>
#include <hip/hip_bf16.h>

// ---------------- problem constants ----------------
#define L_     4096
#define DM_    2048
#define H_     32
#define P_     128
#define INTER_ 4096
#define N_     64
#define KCONV_ 4
#define DTR_   128
#define Q_     256
#define NC_    16      // L/Q chunks

typedef __attribute__((ext_vector_type(16))) _Float16 v16h;
typedef __attribute__((ext_vector_type(8)))  _Float16 h8v;
typedef __attribute__((ext_vector_type(4)))  _Float16 h4v;
typedef __attribute__((ext_vector_type(2)))  _Float16 h2v;
typedef __attribute__((ext_vector_type(8)))  float    v8f;

// ---------------- TDM availability ----------------
#if defined(__has_builtin)
#  if __has_builtin(__builtin_amdgcn_tensor_load_to_lds) && \
      __has_builtin(__builtin_amdgcn_s_wait_tensorcnt)
#    define HAVE_TDM 1
#  endif
#endif
#ifndef HAVE_TDM
#  define HAVE_TDM 0
#endif

#if HAVE_TDM
typedef unsigned int u32x4 __attribute__((ext_vector_type(4)));
typedef int          i32x8 __attribute__((ext_vector_type(8)));
typedef int          i32x4 __attribute__((ext_vector_type(4)));

// Issue one TDM 2-D tile load: 32 (K) x 128 (rows) f16 elements, row stride K,
// packed row-major into LDS at byte offset ldsOff. D# packing per ISA 8.3/8.4.
// 6-arg builtin form on this toolchain.
__device__ inline void tdm_load_tile_f16(const _Float16* g, unsigned ldsOff,
                                         int K) {
  unsigned long long ga = (unsigned long long)(__SIZE_TYPE__)g;
  u32x4 g0;
  g0[0] = 1u;                                   // count=1, user mode
  g0[1] = ldsOff;                               // lds_addr [63:32]
  g0[2] = (unsigned)ga;                         // global_addr lo
  g0[3] = (unsigned)((ga >> 32) & 0x01ffffffull) | 0x80000000u; // ga hi | type=2
  i32x8 g1;
  g1[0] = 1 << 16;                              // data_size = 2 bytes
  g1[1] = (int)((unsigned)K << 16);             // tensor_dim0 lo16 @[63:48]
  g1[2] = (int)(((unsigned)K >> 16) | (128u << 16)); // dim0 hi | tensor_dim1 lo
  g1[3] = (int)(32u << 16);                     // dim1 hi | tile_dim0 = 32
  g1[4] = 128;                                  // tile_dim1 = 128, tile_dim2 = 0
  g1[5] = K;                                    // tensor_dim0_stride lo32
  g1[6] = 0;                                    // stride hi | dim1_stride lo
  g1[7] = 0;
  i32x4 z; z[0] = z[1] = z[2] = z[3] = 0;
  i32x8 z8;
#pragma unroll
  for (int i = 0; i < 8; ++i) z8[i] = 0;
  __builtin_amdgcn_tensor_load_to_lds(g0, g1, z, z, z8, 0);
}
#endif

// ---------------- WMMA helpers ----------------
__device__ inline v8f zero8() {
  v8f z;
#pragma unroll
  for (int i = 0; i < 8; ++i) z[i] = 0.f;
  return z;
}

__device__ inline v8f wmma_f16(v16h a, v16h b, v8f c) {
  return __builtin_amdgcn_wmma_f32_16x16x32_f16(
      false, a, false, b, (short)0, c, false, false);
}

// A fragment: 16x32 f16, M-major LDS tile lds[row][ldm].
__device__ inline v16h load_a_frag(const _Float16* lds, int row0, int ldm,
                                   int k0, int lane) {
  const int m = lane & 15, hi = lane >> 4;
  const _Float16* base = lds + (row0 + m) * ldm + k0 + hi * 8;
  h8v lo = *(const h8v*)base;
  h8v hh = *(const h8v*)(base + 16);
  v16h r;
#pragma unroll
  for (int i = 0; i < 8; ++i) { r[i] = lo[i]; r[i + 8] = hh[i]; }
  return r;
}

// B fragment: 32x16 f16, N-major LDS tile lds[col][ldn].
__device__ inline v16h load_b_frag(const _Float16* lds, int col0, int ldn,
                                   int k0, int lane) {
  const int n = lane & 15, hi = lane >> 4;
  const _Float16* base = lds + (col0 + n) * ldn + k0 + hi * 8;
  h8v lo = *(const h8v*)base;
  h8v hh = *(const h8v*)(base + 16);
  v16h r;
#pragma unroll
  for (int i = 0; i < 8; ++i) { r[i] = lo[i]; r[i + 8] = hh[i]; }
  return r;
}

__device__ inline float silu_f(float x) { return x / (1.f + __expf(-x)); }

// ---------------- f16 GEMM: C[M,N] = A[M,K] @ Bt[N,K]^T ----------------
// A, Bt pre-converted f16 row-major panels. 256 thr, tile 128x128, BK=32.
// TDM path: double-buffered LDS tiles; wave 0 issues the DMA for tile k+1
// while all waves run WMMAs on tile k (DMA hidden behind matrix math).
__global__ __launch_bounds__(256) void gemm_f16(
    const _Float16* __restrict__ A, const _Float16* __restrict__ Bt,
    float* __restrict__ C, int M, int N, int K,
    long sA, long sB, long sC) {
#if HAVE_TDM
  __shared__ _Float16 As[2][128 * 32];
  __shared__ _Float16 Bs[2][128 * 32];
#else
  __shared__ _Float16 As[1][128 * 32];
  __shared__ _Float16 Bs[1][128 * 32];
#endif
  const int tid = threadIdx.x, lane = tid & 31, wave = tid >> 5;
  const int bm = blockIdx.y * 128, bn = blockIdx.x * 128;
  A += (long)blockIdx.z * sA;
  Bt += (long)blockIdx.z * sB;
  C += (long)blockIdx.z * sC;
  const int wm = wave >> 1, wn = wave & 1;

  v8f acc[2][4];
#pragma unroll
  for (int mi = 0; mi < 2; ++mi)
#pragma unroll
    for (int ni = 0; ni < 4; ++ni) acc[mi][ni] = zero8();

  const int nk = K >> 5;  // K/32 tiles

#if HAVE_TDM
  if (tid < 32) {  // prologue: fetch tile 0
    tdm_load_tile_f16(A + (long)bm * K, (unsigned)(__SIZE_TYPE__)(&As[0][0]), K);
    tdm_load_tile_f16(Bt + (long)bn * K, (unsigned)(__SIZE_TYPE__)(&Bs[0][0]), K);
  }
  int cur = 0;
  for (int kk = 0; kk < nk; ++kk) {
    if (tid < 32) __builtin_amdgcn_s_wait_tensorcnt(0);  // tile 'cur' landed
    __syncthreads();
    if (tid < 32 && (kk + 1) < nk) {  // prefetch tile kk+1 into other buffer
      int k1 = (kk + 1) * 32;
      tdm_load_tile_f16(A + (long)bm * K + k1,
                        (unsigned)(__SIZE_TYPE__)(&As[cur ^ 1][0]), K);
      tdm_load_tile_f16(Bt + (long)bn * K + k1,
                        (unsigned)(__SIZE_TYPE__)(&Bs[cur ^ 1][0]), K);
    }
    const _Float16* Ap = &As[cur][0];
    const _Float16* Bp = &Bs[cur][0];
    v16h af0 = load_a_frag(Ap, wm * 32, 32, 0, lane);
    v16h af1 = load_a_frag(Ap, wm * 32 + 16, 32, 0, lane);
#pragma unroll
    for (int ni = 0; ni < 4; ++ni) {
      v16h bf = load_b_frag(Bp, wn * 64 + ni * 16, 32, 0, lane);
      acc[0][ni] = wmma_f16(af0, bf, acc[0][ni]);
      acc[1][ni] = wmma_f16(af1, bf, acc[1][ni]);
    }
    __syncthreads();
    cur ^= 1;
  }
#else
  for (int kk = 0; kk < nk; ++kk) {
    int k0 = kk * 32;
#pragma unroll
    for (int i = 0; i < 2; ++i) {
      int q = tid + 256 * i;
      int r = q >> 2, cg = (q & 3) * 8;
      *(h8v*)&As[0][r * 32 + cg] =
          *(const h8v*)(A + (long)(bm + r) * K + k0 + cg);
    }
#pragma unroll
    for (int i = 0; i < 2; ++i) {
      int q = tid + 256 * i;
      int r = q >> 2, cg = (q & 3) * 8;
      *(h8v*)&Bs[0][r * 32 + cg] =
          *(const h8v*)(Bt + (long)(bn + r) * K + k0 + cg);
    }
    __syncthreads();
    v16h af0 = load_a_frag(&As[0][0], wm * 32, 32, 0, lane);
    v16h af1 = load_a_frag(&As[0][0], wm * 32 + 16, 32, 0, lane);
#pragma unroll
    for (int ni = 0; ni < 4; ++ni) {
      v16h bf = load_b_frag(&Bs[0][0], wn * 64 + ni * 16, 32, 0, lane);
      acc[0][ni] = wmma_f16(af0, bf, acc[0][ni]);
      acc[1][ni] = wmma_f16(af1, bf, acc[1][ni]);
    }
    __syncthreads();
  }
#endif
  const int nl = lane & 15, hi = lane >> 4;
#pragma unroll
  for (int mi = 0; mi < 2; ++mi)
#pragma unroll
    for (int ni = 0; ni < 4; ++ni)
#pragma unroll
      for (int r = 0; r < 8; ++r)
        C[(long)(bm + wm * 32 + mi * 16 + hi * 8 + r) * N +
          bn + wn * 64 + ni * 16 + nl] = acc[mi][ni][r];
}

// ---------------- f32 -> f16 convert (contiguous) ----------------
__global__ __launch_bounds__(256) void cvt_f16_kernel(
    const float* __restrict__ src, _Float16* __restrict__ dst) {
  long q = (long)blockIdx.x * 256 + threadIdx.x;
  float4 v = *(const float4*)(src + q * 4);
  h4v d;
  d[0] = (_Float16)v.x; d[1] = (_Float16)v.y;
  d[2] = (_Float16)v.z; d[3] = (_Float16)v.w;
  *(h4v*)(dst + q * 4) = d;
}

// ---------------- f32 [R,C] -> f16 [C,R] tiled transpose ----------------
__global__ __launch_bounds__(256) void transpose_cvt_kernel(
    const float* __restrict__ src, _Float16* __restrict__ dst, int R, int C) {
  __shared__ _Float16 t[64][65];
  const int c0 = blockIdx.x * 64, r0 = blockIdx.y * 64;
  const int tid = threadIdx.x;
#pragma unroll
  for (int i = 0; i < 16; ++i) {
    int q = tid + 256 * i;
    int r = q >> 6, c = q & 63;
    t[r][c] = (_Float16)src[(long)(r0 + r) * C + c0 + c];
  }
  __syncthreads();
#pragma unroll
  for (int i = 0; i < 16; ++i) {
    int q = tid + 256 * i;
    int c = q >> 6, r = q & 63;
    dst[(long)(c0 + c) * R + r0 + r] = t[r][c];
  }
}

// ---------------- conv(K=4) + SiLU -> f16 ----------------
__global__ __launch_bounds__(256) void conv_silu_kernel(
    const float* __restrict__ proj, const float* __restrict__ conv_w,
    _Float16* __restrict__ xc16) {
  long idx = (long)blockIdx.x * 256 + threadIdx.x;  // L*INTER
  int i = (int)(idx % INTER_);
  long l = idx / INTER_;
  float s = 0.f;
#pragma unroll
  for (int k = 0; k < KCONV_; ++k) {
    long ls = l - (KCONV_ - 1) + k;
    if (ls >= 0) s += proj[ls * (2 * INTER_) + INTER_ + i] * conv_w[i * KCONV_ + k];
  }
  xc16[idx] = (_Float16)silu_f(s);
}

// ---------------- RMSNorm over a column segment ----------------
__global__ void rmsnorm_f32_kernel(const float* __restrict__ src, int rowStride,
                                   int col0, int n, const float* __restrict__ w,
                                   float* __restrict__ dst) {
  __shared__ float red[128];
  int row = blockIdx.x, j = threadIdx.x;
  float v = src[(long)row * rowStride + col0 + j];
  red[j] = v * v;
  __syncthreads();
  for (int s = n >> 1; s > 0; s >>= 1) {
    if (j < s) red[j] += red[j + s];
    __syncthreads();
  }
  float rms = rsqrtf(red[0] / (float)n + 1e-6f);
  dst[(long)row * n + j] = v * rms * w[j];
}

__global__ void rmsnorm_f16_kernel(const float* __restrict__ src, int rowStride,
                                   int col0, int n, const float* __restrict__ w,
                                   _Float16* __restrict__ dst) {
  __shared__ float red[128];
  int row = blockIdx.x, j = threadIdx.x;
  float v = src[(long)row * rowStride + col0 + j];
  red[j] = v * v;
  __syncthreads();
  for (int s = n >> 1; s > 0; s >>= 1) {
    if (j < s) red[j] += red[j + s];
    __syncthreads();
  }
  float rms = rsqrtf(red[0] / (float)n + 1e-6f);
  dst[(long)row * n + j] = (_Float16)(v * rms * w[j]);
}

// ---------------- dt = softplus(tsn @ Wdt + bias) ----------------
__global__ __launch_bounds__(256) void dtproj_kernel(
    const float* __restrict__ tsn, const float* __restrict__ Wdt,
    const float* __restrict__ bias, float* __restrict__ dt) {
  int idx = blockIdx.x * 256 + threadIdx.x;  // L*H
  int h = idx & (H_ - 1);
  int l = idx >> 5;
  float s = bias[h];
#pragma unroll 4
  for (int d = 0; d < DTR_; ++d) s += tsn[(long)l * DTR_ + d] * Wdt[d * H_ + h];
  dt[idx] = (s > 20.f) ? s : log1pf(__expf(s));
}

// ---------------- seg = per-chunk cumsum of dt*A ----------------
__global__ __launch_bounds__(256) void seg_kernel(
    const float* __restrict__ dt, const float* __restrict__ A,
    float* __restrict__ seg) {
  int tid = threadIdx.x, lane = tid & 31, wave = tid >> 5;
  int pair = blockIdx.x * 8 + wave;  // NC_*H_ pairs
  int c = pair >> 5, h = pair & 31;
  float a = A[h];
  float v[8];
  float s = 0.f;
  int base = c * Q_ + lane * 8;
#pragma unroll
  for (int i = 0; i < 8; ++i) {
    v[i] = dt[(long)(base + i) * H_ + h] * a;
    s += v[i];
  }
  float tot = s;
#pragma unroll
  for (int off = 1; off < 32; off <<= 1) {
    float nv = __shfl_up(tot, off, 32);
    if (lane >= off) tot += nv;
  }
  float run = tot - s;
#pragma unroll
  for (int i = 0; i < 8; ++i) {
    run += v[i];
    seg[(long)(base + i) * H_ + h] = run;
  }
}

// ---------------- Yd: intra-chunk masked attention via WMMA ----------------
__global__ __launch_bounds__(256) void yd_kernel(
    const float* __restrict__ CB, const float* __restrict__ seg,
    const float* __restrict__ dt, const _Float16* __restrict__ xc16,
    float* __restrict__ yAcc) {
  __shared__ float segs[Q_], dts[Q_];
  __shared__ _Float16 Ms[64 * 32];    // A operand: M[l][s]
  __shared__ _Float16 Xs[128 * 32];   // B operand, N-major: X[p][s]
  const int lt = blockIdx.x, h = blockIdx.y, c = blockIdx.z;
  const int tid = threadIdx.x, lane = tid & 31, wave = tid >> 5;
  segs[tid] = seg[((long)c * Q_ + tid) * H_ + h];
  dts[tid]  = dt[((long)c * Q_ + tid) * H_ + h];
  __syncthreads();
  const int l0 = lt * 64;
  const int wl = wave >> 1, wp = wave & 1;
  v8f acc[4];
#pragma unroll
  for (int ni = 0; ni < 4; ++ni) acc[ni] = zero8();

  const int scp = (tid & 15) * 2;  // thread-invariant s pair within tile
  const int nst = lt * 2 + 2;
  for (int st = 0; st < nst; ++st) {
    int s0 = st * 32;
    int sg0 = s0 + scp, sg1 = sg0 + 1;
    float segS0 = segs[sg0], segS1 = segs[sg1];
    float dS0 = dts[sg0], dS1 = dts[sg1];
    // stage M tile 64x32, branchless, packed 2-half stores
#pragma unroll
    for (int i = 0; i < 4; ++i) {
      int r = (tid >> 4) + 16 * i;
      int lg = l0 + r;
      float2 cb = *(const float2*)&CB[((long)c * Q_ + lg) * Q_ + sg0];
      float segL = segs[lg];
      float m0 = (sg0 <= lg) ? cb.x * __expf(segL - segS0) * dS0 : 0.f;
      float m1 = (sg1 <= lg) ? cb.y * __expf(segL - segS1) * dS1 : 0.f;
      h2v pk; pk[0] = (_Float16)m0; pk[1] = (_Float16)m1;
      *(h2v*)&Ms[r * 32 + scp] = pk;
    }
    // stage X tile (32 s) x (128 p) transposed, raw f16
#pragma unroll
    for (int i = 0; i < 16; ++i) {
      int q = tid + 256 * i;
      int sr = q >> 7, p = q & 127;
      Xs[p * 32 + sr] = xc16[((long)c * Q_ + s0 + sr) * INTER_ + h * P_ + p];
    }
    __syncthreads();
    v16h af = load_a_frag(Ms, wl * 16, 32, 0, lane);
#pragma unroll
    for (int ni = 0; ni < 4; ++ni) {
      v16h bf = load_b_frag(Xs, wp * 64 + ni * 16, 32, 0, lane);
      acc[ni] = wmma_f16(af, bf, acc[ni]);
    }
    __syncthreads();
  }
  const int nl = lane & 15, hi = lane >> 4;
#pragma unroll
  for (int ni = 0; ni < 4; ++ni)
#pragma unroll
    for (int r = 0; r < 8; ++r) {
      long l = (long)c * Q_ + l0 + wl * 16 + hi * 8 + r;
      int col = h * P_ + wp * 64 + ni * 16 + nl;
      yAcc[l * INTER_ + col] = acc[ni][r];
    }
}

// ---------------- states[c,h,p,n] ----------------
__global__ __launch_bounds__(256) void states_kernel(
    const _Float16* __restrict__ xc16, const _Float16* __restrict__ Bn16,
    const float* __restrict__ seg, const float* __restrict__ dt,
    float* __restrict__ states) {
  __shared__ float dd[Q_];
  __shared__ _Float16 At[128 * 32];
  __shared__ _Float16 Bt[64 * 32];
  const int h = blockIdx.x, c = blockIdx.y;
  const int tid = threadIdx.x, lane = tid & 31, wave = tid >> 5;
  float seglast = seg[((long)c * Q_ + (Q_ - 1)) * H_ + h];
  {
    float sg = seg[((long)c * Q_ + tid) * H_ + h];
    dd[tid] = dt[((long)c * Q_ + tid) * H_ + h] * __expf(seglast - sg);
  }
  __syncthreads();
  v8f acc[4];
#pragma unroll
  for (int ni = 0; ni < 4; ++ni) acc[ni] = zero8();

  for (int s0 = 0; s0 < Q_; s0 += 32) {
#pragma unroll
    for (int i = 0; i < 16; ++i) {
      int q = tid + 256 * i;
      int sr = q >> 7, p = q & 127;
      At[p * 32 + sr] = (_Float16)(
          (float)xc16[((long)c * Q_ + s0 + sr) * INTER_ + h * P_ + p] * dd[s0 + sr]);
    }
#pragma unroll
    for (int i = 0; i < 8; ++i) {
      int q = tid + 256 * i;
      int sr = q >> 6, n = q & 63;
      Bt[n * 32 + sr] = Bn16[((long)c * Q_ + s0 + sr) * N_ + n];
    }
    __syncthreads();
    v16h af = load_a_frag(At, wave * 16, 32, 0, lane);
#pragma unroll
    for (int ni = 0; ni < 4; ++ni) {
      v16h bf = load_b_frag(Bt, ni * 16, 32, 0, lane);
      acc[ni] = wmma_f16(af, bf, acc[ni]);
    }
    __syncthreads();
  }
  const int nl = lane & 15, hi = lane >> 4;
#pragma unroll
  for (int ni = 0; ni < 4; ++ni)
#pragma unroll
    for (int r = 0; r < 8; ++r) {
      int p = wave * 16 + hi * 8 + r;
      int n = ni * 16 + nl;
      states[(((long)c * H_ + h) * P_ + p) * N_ + n] = acc[ni][r];
    }
}

// ---------------- serial inter-chunk scan -> prevs (f16) ----------------
__global__ __launch_bounds__(256) void scan_kernel(
    const float* __restrict__ states, const float* __restrict__ seg,
    _Float16* __restrict__ prevs16) {
  long idx = (long)blockIdx.x * 256 + threadIdx.x;  // H*P*N
  int n = (int)(idx & 63);
  int p = (int)((idx >> 6) & 127);
  int h = (int)(idx >> 13);
  float run = 0.f;
  for (int c = 0; c < NC_; ++c) {
    long off = (((long)c * H_ + h) * P_ + p) * N_ + n;
    prevs16[off] = (_Float16)run;
    float cd = __expf(seg[((long)c * Q_ + (Q_ - 1)) * H_ + h]);
    run = run * cd + states[off];
  }
}

// ---------------- Yo = exp(seg)*C@prevs^T + epilogue -> y16 ----------------
__global__ __launch_bounds__(256) void yo_combine_kernel(
    const _Float16* __restrict__ Cn16, const _Float16* __restrict__ prevs16,
    const float* __restrict__ seg, const _Float16* __restrict__ xc16,
    const float* __restrict__ proj, const float* __restrict__ Dv,
    const float* __restrict__ yAcc, _Float16* __restrict__ y16) {
  __shared__ _Float16 As[64 * 64];    // C[l][n]
  __shared__ _Float16 Bt[128 * 64];   // prevs[p][n] (already B^T layout)
  const int lt = blockIdx.x, h = blockIdx.y, c = blockIdx.z;
  const int tid = threadIdx.x, lane = tid & 31, wave = tid >> 5;
#pragma unroll
  for (int i = 0; i < 2; ++i) {       // 64x64 halves, raw 16B copies
    int q = tid + 256 * i;
    int r = q >> 3, cg = (q & 7) * 8;
    *(h8v*)&As[r * 64 + cg] =
        *(const h8v*)&Cn16[((long)c * Q_ + lt * 64 + r) * N_ + cg];
  }
#pragma unroll
  for (int i = 0; i < 4; ++i) {       // 128x64 halves, raw 16B copies
    int q = tid + 256 * i;
    int p = q >> 3, cg = (q & 7) * 8;
    *(h8v*)&Bt[p * 64 + cg] =
        *(const h8v*)&prevs16[(((long)c * H_ + h) * P_ + p) * N_ + cg];
  }
  __syncthreads();
  const int wl = wave & 3, wp = wave >> 2;
  v8f acc[4];
#pragma unroll
  for (int ni = 0; ni < 4; ++ni) acc[ni] = zero8();
#pragma unroll
  for (int k0 = 0; k0 < 64; k0 += 32) {
    v16h af = load_a_frag(As, wl * 16, 64, k0, lane);
#pragma unroll
    for (int ni = 0; ni < 4; ++ni) {
      v16h bf = load_b_frag(Bt, wp * 64 + ni * 16, 64, k0, lane);
      acc[ni] = wmma_f16(af, bf, acc[ni]);
    }
  }
  const int nl = lane & 15, hi = lane >> 4;
  const float dh = Dv[h];
#pragma unroll
  for (int ni = 0; ni < 4; ++ni)
#pragma unroll
    for (int r = 0; r < 8; ++r) {
      long l = (long)c * Q_ + lt * 64 + wl * 16 + hi * 8 + r;
      int col = h * P_ + wp * 64 + ni * 16 + nl;
      float e = __expf(seg[l * H_ + h]);
      float xv = (float)xc16[l * INTER_ + col];
      float g = proj[l * (2 * INTER_) + col];  // gate = first INTER cols
      float yv = yAcc[l * INTER_ + col] + e * acc[ni][r] + dh * xv;
      y16[l * INTER_ + col] = (_Float16)(yv * silu_f(g));
    }
}

// ---------------- host orchestration ----------------
extern "C" void kernel_launch(void* const* d_in, const int* in_sizes, int n_in,
                              void* d_out, int out_size, void* d_ws,
                              size_t ws_size, hipStream_t stream) {
  const float* hidden    = (const float*)d_in[0];
  const float* in_proj   = (const float*)d_in[1];
  const float* conv_w    = (const float*)d_in[2];
  const float* bcdt_w    = (const float*)d_in[3];
  const float* dt_norm_w = (const float*)d_in[4];
  const float* B_norm_w  = (const float*)d_in[5];
  const float* C_norm_w  = (const float*)d_in[6];
  const float* dt_proj_w = (const float*)d_in[7];
  const float* Av        = (const float*)d_in[8];
  const float* Dv        = (const float*)d_in[9];
  const float* dt_bias   = (const float*)d_in[10];
  const float* out_proj  = (const float*)d_in[11];
  float* out = (float*)d_out;

  char* ws = (char*)d_ws;
  float*    proj    = (float*)ws;    ws += (size_t)L_ * 2 * INTER_ * 4;
  float*    yAcc    = (float*)ws;    ws += (size_t)L_ * INTER_ * 4;
  float*    bcdt    = (float*)ws;    ws += (size_t)L_ * 256 * 4;
  float*    tsn     = (float*)ws;    ws += (size_t)L_ * DTR_ * 4;
  float*    dtb     = (float*)ws;    ws += (size_t)L_ * H_ * 4;
  float*    segb    = (float*)ws;    ws += (size_t)L_ * H_ * 4;
  float*    CBb     = (float*)ws;    ws += (size_t)NC_ * Q_ * Q_ * 4;
  float*    statesb = (float*)ws;    ws += (size_t)NC_ * H_ * P_ * N_ * 4;
  _Float16* hidden16= (_Float16*)ws; ws += (size_t)L_ * DM_ * 2;
  _Float16* wInT    = (_Float16*)ws; ws += (size_t)DM_ * 2 * INTER_ * 2;
  _Float16* wBcdtT  = (_Float16*)ws; ws += (size_t)INTER_ * 256 * 2;
  _Float16* wOutT   = (_Float16*)ws; ws += (size_t)INTER_ * DM_ * 2;
  _Float16* xc16    = (_Float16*)ws; ws += (size_t)L_ * INTER_ * 2;
  _Float16* y16     = (_Float16*)ws; ws += (size_t)L_ * INTER_ * 2;
  _Float16* Bn16    = (_Float16*)ws; ws += (size_t)L_ * N_ * 2;
  _Float16* Cn16    = (_Float16*)ws; ws += (size_t)L_ * N_ * 2;
  _Float16* prevs16 = (_Float16*)ws; ws += (size_t)NC_ * H_ * P_ * N_ * 2;

  // 0) pre-convert / pre-transpose operands to f16 panels
  cvt_f16_kernel<<<(L_ * DM_) / 1024, 256, 0, stream>>>(hidden, hidden16);
  transpose_cvt_kernel<<<dim3(2 * INTER_ / 64, DM_ / 64), 256, 0, stream>>>(
      in_proj, wInT, DM_, 2 * INTER_);
  transpose_cvt_kernel<<<dim3(256 / 64, INTER_ / 64), 256, 0, stream>>>(
      bcdt_w, wBcdtT, INTER_, 256);
  transpose_cvt_kernel<<<dim3(DM_ / 64, INTER_ / 64), 256, 0, stream>>>(
      out_proj, wOutT, INTER_, DM_);

  // 1) proj = hidden @ in_proj_w
  gemm_f16<<<dim3(2 * INTER_ / 128, L_ / 128, 1), 256, 0, stream>>>(
      hidden16, wInT, proj, L_, 2 * INTER_, DM_, 0, 0, 0);

  // 2) causal conv + SiLU -> xc16
  conv_silu_kernel<<<(L_ * INTER_) / 256, 256, 0, stream>>>(proj, conv_w, xc16);

  // 3) bcdt = xc @ bcdt_w
  gemm_f16<<<dim3(256 / 128, L_ / 128, 1), 256, 0, stream>>>(
      xc16, wBcdtT, bcdt, L_, 2 * N_ + DTR_, INTER_, 0, 0, 0);

  // 4) RMS norms
  rmsnorm_f16_kernel<<<L_, N_, 0, stream>>>(bcdt, 256, 0, N_, B_norm_w, Bn16);
  rmsnorm_f16_kernel<<<L_, N_, 0, stream>>>(bcdt, 256, N_, N_, C_norm_w, Cn16);
  rmsnorm_f32_kernel<<<L_, DTR_, 0, stream>>>(bcdt, 256, 2 * N_, DTR_, dt_norm_w, tsn);

  // 5) dt, 6) seg
  dtproj_kernel<<<(L_ * H_) / 256, 256, 0, stream>>>(tsn, dt_proj_w, dt_bias, dtb);
  seg_kernel<<<(NC_ * H_) / 8, 256, 0, stream>>>(dtb, Av, segb);

  // 7) CB[c] = C_chunk @ B_chunk^T (batched)
  gemm_f16<<<dim3(2, 2, NC_), 256, 0, stream>>>(
      Cn16, Bn16, CBb, Q_, Q_, N_, (long)Q_ * N_, (long)Q_ * N_, (long)Q_ * Q_);

  // 8) Yd, 9) states + scan + Yo/epilogue
  yd_kernel<<<dim3(4, H_, NC_), 256, 0, stream>>>(CBb, segb, dtb, xc16, yAcc);
  states_kernel<<<dim3(H_, NC_), 256, 0, stream>>>(xc16, Bn16, segb, dtb, statesb);
  scan_kernel<<<(H_ * P_ * N_) / 256, 256, 0, stream>>>(statesb, segb, prevs16);
  yo_combine_kernel<<<dim3(4, H_, NC_), 256, 0, stream>>>(
      Cn16, prevs16, segb, xc16, proj, Dv, yAcc, y16);

  // 10) out = y @ out_proj_w
  gemm_f16<<<dim3(DM_ / 128, L_ / 128, 1), 256, 0, stream>>>(
      y16, wOutT, out, L_, DM_, INTER_, 0, 0, 0);
}
